// TwoModelBlockRouted_78271484002775
// MI455X (gfx1250) — matrix operand order repaired and checked by reading, not compiled, not run on hardware
//
#include <hip/hip_runtime.h>
#include <stdint.h>

#define M_TOTAL 4096
#define HDIM    2048
#define NHEADS  16
#define HEADD   128
#define FFD     5504

typedef __attribute__((ext_vector_type(16))) __bf16 v16bf;
typedef __attribute__((ext_vector_type(8)))  float  v8f;
typedef int v4i_ __attribute__((vector_size(16)));   // b128 payload type

union ABf { v16bf v; unsigned short u[16]; uint4 q[2]; };

__device__ __forceinline__ unsigned short f2bf(float f) {
  union { float f; uint32_t u; } c; c.f = f;
  uint32_t u = c.u;
  u += 0x7FFFu + ((u >> 16) & 1u);   // round-to-nearest-even bf16
  return (unsigned short)(u >> 16);
}
__device__ __forceinline__ float bf2f(unsigned short h) {
  union { uint32_t u; float f; } c; c.u = ((uint32_t)h) << 16;
  return c.f;
}
__device__ __forceinline__ v8f vzero8() {
  v8f v;
#pragma unroll
  for (int i = 0; i < 8; ++i) v[i] = 0.0f;
  return v;
}
__device__ __forceinline__ v8f wmma_bf(v16bf a, v16bf b, v8f c) {
  return __builtin_amdgcn_wmma_f32_16x16x32_bf16(false, a, false, b, (short)0, c,
                                                 false, false);
}

// --------------------- async global->LDS copy (16B/lane) --------------------
#if __has_builtin(__builtin_amdgcn_global_load_async_to_lds_b128)
#define HAS_ASYNC_LDS 1
__device__ __forceinline__ void async_cp16(const void* g, void* l) {
  __builtin_amdgcn_global_load_async_to_lds_b128(
      (__attribute__((address_space(1))) v4i_*)(void*)g,
      (__attribute__((address_space(3))) v4i_*)l, 0, 0);
}
__device__ __forceinline__ void async_wait0() {
#if __has_builtin(__builtin_amdgcn_s_wait_asynccnt)
  __builtin_amdgcn_s_wait_asynccnt(0);
#else
  asm volatile("s_wait_asynccnt 0" ::: "memory");
#endif
}
#else
#define HAS_ASYNC_LDS 0
__device__ __forceinline__ void async_cp16(const void* g, void* l) {
  *(uint4*)l = *(const uint4*)g;
}
__device__ __forceinline__ void async_wait0() {}
#endif

// ---------------------------------------------------------------------------
// Weight convert+transpose: fp32 W[model][K,N] -> bf16 Wt[model][N,K]
// 32x32 tiles through LDS so both global sides stay coalesced.
// ---------------------------------------------------------------------------
__global__ __launch_bounds__(256) void transpose_bf16_k(
    const float* __restrict__ W, unsigned short* __restrict__ Wt, int K, int N) {
  __shared__ unsigned short tile[32][33];
  const float* Wm = W + (size_t)blockIdx.z * (size_t)K * N;
  unsigned short* Wtm = Wt + (size_t)blockIdx.z * (size_t)K * N;
  const int nb = blockIdx.x * 32, kb = blockIdx.y * 32;
  const int tx = threadIdx.x, ty = threadIdx.y;  // 32 x 8
#pragma unroll
  for (int i = 0; i < 32; i += 8)
    tile[ty + i][tx] = f2bf(Wm[(size_t)(kb + ty + i) * N + nb + tx]);
  __syncthreads();
#pragma unroll
  for (int i = 0; i < 32; i += 8)
    Wtm[(size_t)(nb + ty + i) * K + kb + tx] = tile[tx][ty + i];
}

// ---------------------------------------------------------------------------
// RMSNorm: fp32 [M,H] -> bf16 [M,H], routed LN weight by split point.
// ---------------------------------------------------------------------------
__global__ __launch_bounds__(256) void rmsnorm_bf16_k(
    const float* __restrict__ x, const float* __restrict__ ln,
    const int* __restrict__ sp, unsigned short* __restrict__ out) {
  __shared__ float red[8];
  const int row = blockIdx.x;
  const int s = *sp;
  const float* lw = ln + ((row < s) ? 0 : HDIM);
  const float* xr = x + (size_t)row * HDIM;
  float acc = 0.0f;
  for (int c = threadIdx.x; c < HDIM; c += 256) {
    float v = xr[c];
    acc += v * v;
  }
#pragma unroll
  for (int m = 1; m < 32; m <<= 1) acc += __shfl_xor(acc, m, 32);
  if ((threadIdx.x & 31) == 0) red[threadIdx.x >> 5] = acc;
  __syncthreads();
  float tot = 0.0f;
#pragma unroll
  for (int j = 0; j < 8; ++j) tot += red[j];
  const float rs = rsqrtf(tot * (1.0f / (float)HDIM) + 1.1920929e-07f);
  for (int c = threadIdx.x; c < HDIM; c += 256)
    out[(size_t)row * HDIM + c] = f2bf(xr[c] * rs * lw[c]);
}

// ---------------------------------------------------------------------------
// Routed GEMM:  C[M,N] = A[M,K](bf16) * Wt[model][N,K](bf16, pre-transposed)
// 128x128 tile, BK=32, 8 waves (4x2), bf16 WMMA. Double-buffered LDS tiles
// filled entirely with global_load_async_to_lds_b128.
// ---------------------------------------------------------------------------
#define BM 128
#define BN 128
#define BK 32
#define LDA 40   // padded ushort stride (80B, 16B-aligned)
#define LDB 40

template <bool OUT_BF16, bool ADD_RES>
__global__ __launch_bounds__(256) void gemm_routed_k(
    const unsigned short* __restrict__ A, const unsigned short* __restrict__ Wt,
    const float* __restrict__ Res, float* __restrict__ Cf,
    unsigned short* __restrict__ Cbf, const int* __restrict__ sp, int N, int K) {
  __shared__ unsigned short As[2][BM * LDA];
  __shared__ unsigned short Bs[2][BN * LDB];  // N-major: Bs[n][k]

  const int s = *sp;
  const int m0 = blockIdx.x * BM;
  const int n0 = blockIdx.y * BN;
  const unsigned short* Wm =
      Wt + (size_t)((m0 >= s) ? 1 : 0) * (size_t)K * (size_t)N;

  const int t = threadIdx.x;
  const int lane = t & 31;
  const int w = t >> 5;
  const int wm = w >> 1;   // 0..3 -> rows wm*32
  const int wn = w & 1;    // 0..1 -> cols wn*64
  const int half = lane >> 4;
  const int l16 = lane & 15;

  auto issue_tile = [&](int buf, int k0) {
#pragma unroll
    for (int it = 0; it < 2; ++it) {   // A: 128x32 bf16 = 512 x b128
      int idx = t + it * 256;
      int r = idx >> 2, cc = idx & 3;
      async_cp16(A + (size_t)(m0 + r) * K + k0 + cc * 8,
                 &As[buf][r * LDA + cc * 8]);
    }
#pragma unroll
    for (int it = 0; it < 2; ++it) {   // B: 128 n-rows x 32 k = 512 x b128
      int idx = t + it * 256;
      int nn = idx >> 2, cc = idx & 3;
      async_cp16(Wm + (size_t)(n0 + nn) * K + k0 + cc * 8,
                 &Bs[buf][nn * LDB + cc * 8]);
    }
  };

  v8f acc[2][4];
#pragma unroll
  for (int mt = 0; mt < 2; ++mt)
#pragma unroll
    for (int nt = 0; nt < 4; ++nt) acc[mt][nt] = vzero8();

  issue_tile(0, 0);
  int buf = 0;
  for (int k0 = 0; k0 < K; k0 += BK) {
    async_wait0();
    __syncthreads();
    if (k0 + BK < K) issue_tile(buf ^ 1, k0 + BK);
    if (k0 + 2 * BK < K) {  // prefetch k+2 weight slab toward L2
      int pn = t >> 1, pc = (t & 1) << 4;
      __builtin_prefetch(Wm + (size_t)(n0 + pn) * K + k0 + 2 * BK + pc, 0, 0);
    }

    ABf af[2], bw[4];
#pragma unroll
    for (int mt = 0; mt < 2; ++mt) {
      int row = wm * 32 + mt * 16 + l16;
      af[mt].q[0] = *(const uint4*)(&As[buf][row * LDA + 8 * half]);
      af[mt].q[1] = *(const uint4*)(&As[buf][row * LDA + 16 + 8 * half]);
    }
#pragma unroll
    for (int nt = 0; nt < 4; ++nt) {
      int col = wn * 64 + nt * 16 + l16;
      const uint4* p = (const uint4*)(&Bs[buf][col * LDB + 8 * half]);
      bw[nt].q[0] = p[0];
      bw[nt].q[1] = p[2];
    }
#pragma unroll
    for (int mt = 0; mt < 2; ++mt)
#pragma unroll
      for (int nt = 0; nt < 4; ++nt)
        acc[mt][nt] = wmma_bf(af[mt].v, bw[nt].v, acc[mt][nt]);
    buf ^= 1;
  }

#pragma unroll
  for (int mt = 0; mt < 2; ++mt)
#pragma unroll
    for (int nt = 0; nt < 4; ++nt)
#pragma unroll
      for (int v = 0; v < 8; ++v) {
        int row = m0 + wm * 32 + mt * 16 + v + 8 * half;
        int col = n0 + wn * 64 + nt * 16 + l16;
        float val = acc[mt][nt][v];
        if (ADD_RES) val += Res[(size_t)row * N + col];
        if (OUT_BF16)
          Cbf[(size_t)row * N + col] = f2bf(val);
        else
          Cf[(size_t)row * N + col] = val;
      }
}

// ---------------------------------------------------------------------------
// Flash attention per (segment, head, 64-row q tile). 4 waves, 16 q rows each.
// K tile fills are async global->LDS copies; V is transposed on the sync path.
// ---------------------------------------------------------------------------
__global__ __launch_bounds__(128) void flash_attn_k(
    const unsigned short* __restrict__ qkv,  // [M, 3H] bf16
    unsigned short* __restrict__ attn,       // [M, H] bf16
    const int* __restrict__ sp) {
  __shared__ unsigned short Qs[64 * 128];
  __shared__ unsigned short Ks[64 * 128];
  __shared__ unsigned short Vt[128 * 64];
  __shared__ unsigned short Ps[4][16 * 64];

  const int s = *sp;
  const int seg = blockIdx.z;
  const int head = blockIdx.y;
  const int seg0 = seg ? s : 0;
  const int segN = seg ? (M_TOTAL - s) : s;
  const int q0 = blockIdx.x * 64;
  if (q0 >= segN) return;

  const int t = threadIdx.x;
  const int lane = t & 31;
  const int w = t >> 5;
  const int half = lane >> 4;
  const int l16 = lane & 15;
  const float scale = 0.08838834764831845f;  // 1/sqrt(128)

  for (int e = t; e < 64 * 128; e += 128) {
    int r = e >> 7, d = e & 127;
    float qv = bf2f(qkv[(size_t)(seg0 + q0 + r) * (3 * HDIM) + head * HEADD + d]);
    Qs[e] = f2bf(qv * scale);
  }

  v8f O[8];
#pragma unroll
  for (int dt = 0; dt < 8; ++dt) O[dt] = vzero8();
  float mrow[8], lrow[8];
#pragma unroll
  for (int v = 0; v < 8; ++v) { mrow[v] = -1e30f; lrow[v] = 0.0f; }

  for (int k0 = 0; k0 < segN; k0 += 64) {
    __syncthreads();
    // async copy K tile (64 rows x 128 bf16 = 1024 x b128)
#pragma unroll
    for (int it = 0; it < 8; ++it) {
      int idx = t + it * 128;
      int r = idx >> 4, cc = idx & 15;
      async_cp16(qkv + (size_t)(seg0 + k0 + r) * (3 * HDIM) + HDIM +
                     head * HEADD + cc * 8,
                 &Ks[r * 128 + cc * 8]);
    }
    // sync fill transposed V
    for (int e = t; e < 64 * 128; e += 128) {
      int r = e >> 7, d = e & 127;
      Vt[d * 64 + r] =
          qkv[(size_t)(seg0 + k0 + r) * (3 * HDIM) + 2 * HDIM + head * HEADD + d];
    }
    async_wait0();
    __syncthreads();

    // S[16,64] = Q[16,128] * K^T
    v8f S[4];
#pragma unroll
    for (int nt = 0; nt < 4; ++nt) S[nt] = vzero8();
#pragma unroll
    for (int kk = 0; kk < 4; ++kk) {
      ABf aq;
      int row = w * 16 + l16;
      aq.q[0] = *(const uint4*)(&Qs[row * 128 + kk * 32 + 8 * half]);
      aq.q[1] = *(const uint4*)(&Qs[row * 128 + kk * 32 + 16 + 8 * half]);
#pragma unroll
      for (int nt = 0; nt < 4; ++nt) {
        ABf bk;
        int col = nt * 16 + l16;
        const uint4* p = (const uint4*)(&Ks[col * 128 + kk * 32 + 8 * half]);
        bk.q[0] = p[0];
        bk.q[1] = p[2];
        S[nt] = wmma_bf(aq.v, bk.v, S[nt]);
      }
    }

    // online softmax
    float alpha[8];
#pragma unroll
    for (int v = 0; v < 8; ++v) {
      float rm = fmaxf(fmaxf(S[0][v], S[1][v]), fmaxf(S[2][v], S[3][v]));
      rm = fmaxf(rm, __shfl_xor(rm, 1, 32));
      rm = fmaxf(rm, __shfl_xor(rm, 2, 32));
      rm = fmaxf(rm, __shfl_xor(rm, 4, 32));
      rm = fmaxf(rm, __shfl_xor(rm, 8, 32));
      float mnew = fmaxf(mrow[v], rm);
      alpha[v] = __expf(mrow[v] - mnew);
      mrow[v] = mnew;
    }
    float rsum[8];
#pragma unroll
    for (int v = 0; v < 8; ++v) rsum[v] = 0.0f;
#pragma unroll
    for (int nt = 0; nt < 4; ++nt)
#pragma unroll
      for (int v = 0; v < 8; ++v) {
        float p = __expf(S[nt][v] - mrow[v]);
        S[nt][v] = p;
        rsum[v] += p;
      }
#pragma unroll
    for (int v = 0; v < 8; ++v) {
      float rs = rsum[v];
      rs += __shfl_xor(rs, 1, 32);
      rs += __shfl_xor(rs, 2, 32);
      rs += __shfl_xor(rs, 4, 32);
      rs += __shfl_xor(rs, 8, 32);
      lrow[v] = lrow[v] * alpha[v] + rs;
    }
#pragma unroll
    for (int dt = 0; dt < 8; ++dt)
#pragma unroll
      for (int v = 0; v < 8; ++v) O[dt][v] *= alpha[v];

    // stage P (C-layout -> A-layout via LDS)
#pragma unroll
    for (int nt = 0; nt < 4; ++nt)
#pragma unroll
      for (int v = 0; v < 8; ++v)
        Ps[w][(v + 8 * half) * 64 + nt * 16 + l16] = f2bf(S[nt][v]);
    __syncthreads();

    // O[16,128] += P[16,64] * V[64,128]
#pragma unroll
    for (int kk = 0; kk < 2; ++kk) {
      ABf ap;
      ap.q[0] = *(const uint4*)(&Ps[w][l16 * 64 + kk * 32 + 8 * half]);
      ap.q[1] = *(const uint4*)(&Ps[w][l16 * 64 + kk * 32 + 16 + 8 * half]);
#pragma unroll
      for (int dt = 0; dt < 8; ++dt) {
        ABf bv;
        const uint4* p =
            (const uint4*)(&Vt[(dt * 16 + l16) * 64 + kk * 32 + 8 * half]);
        bv.q[0] = p[0];
        bv.q[1] = p[2];
        O[dt] = wmma_bf(ap.v, bv.v, O[dt]);
      }
    }
  }

#pragma unroll
  for (int v = 0; v < 8; ++v) lrow[v] = 1.0f / lrow[v];
#pragma unroll
  for (int dt = 0; dt < 8; ++dt)
#pragma unroll
    for (int v = 0; v < 8; ++v) {
      int row = seg0 + q0 + w * 16 + v + 8 * half;
      int col = head * HEADD + dt * 16 + l16;
      attn[(size_t)row * HDIM + col] = f2bf(O[dt][v] * lrow[v]);
    }
}

// ---------------------------------------------------------------------------
// SwiGLU: gu fp32 [M, 2*FF] -> silu(gate)*up bf16 [M, FF]
// ---------------------------------------------------------------------------
__global__ __launch_bounds__(256) void swiglu_k(const float* __restrict__ gu,
                                                unsigned short* __restrict__ out) {
  size_t idx = (size_t)blockIdx.x * 256 + threadIdx.x;
  size_t total = (size_t)M_TOTAL * FFD;
  if (idx >= total) return;
  size_t r = idx / FFD, c = idx - r * FFD;
  float g = gu[r * (2 * FFD) + c];
  float u = gu[r * (2 * FFD) + FFD + c];
  float sg = g / (1.0f + __expf(-g));
  out[idx] = f2bf(sg * u);
}

// ---------------------------------------------------------------------------
extern "C" void kernel_launch(void* const* d_in, const int* in_sizes, int n_in,
                              void* d_out, int out_size, void* d_ws,
                              size_t ws_size, hipStream_t stream) {
  (void)in_sizes; (void)n_in; (void)out_size; (void)ws_size;
  const float* x     = (const float*)d_in[0];
  const float* W_qkv = (const float*)d_in[1];
  const float* W_o   = (const float*)d_in[2];
  const float* W_gu  = (const float*)d_in[3];
  const float* W_d   = (const float*)d_in[4];
  const float* ln1   = (const float*)d_in[5];
  const float* ln2   = (const float*)d_in[6];
  const int*   sp    = (const int*)d_in[7];

  // workspace layout
  unsigned short* h_bf    = (unsigned short*)d_ws;                  // M*H
  unsigned short* qkv_bf  = h_bf + (size_t)M_TOTAL * HDIM;          // M*3H
  unsigned short* attn_bf = qkv_bf + (size_t)M_TOTAL * 3 * HDIM;    // M*H
  unsigned short* hmlp_bf = attn_bf + (size_t)M_TOTAL * HDIM;       // M*FF
  float* x2 = (float*)(hmlp_bf + (size_t)M_TOTAL * FFD);            // M*H
  float* gu = x2 + (size_t)M_TOTAL * HDIM;                          // M*2FF
  unsigned short* wt_qkv = (unsigned short*)(gu + (size_t)M_TOTAL * 2 * FFD);
  unsigned short* wt_o   = wt_qkv + (size_t)2 * HDIM * 3 * HDIM;    // [2][N][K]
  unsigned short* wt_gu  = wt_o + (size_t)2 * HDIM * HDIM;
  unsigned short* wt_d   = wt_gu + (size_t)2 * HDIM * 2 * FFD;
  float* outf = (float*)d_out;

  dim3 blk256(256), blk128(128), blkT(32, 8);

  // 0) weight convert+transpose fp32[K,N] -> bf16[N,K]
  transpose_bf16_k<<<dim3((3 * HDIM) / 32, HDIM / 32, 2), blkT, 0, stream>>>(
      W_qkv, wt_qkv, HDIM, 3 * HDIM);
  transpose_bf16_k<<<dim3(HDIM / 32, HDIM / 32, 2), blkT, 0, stream>>>(
      W_o, wt_o, HDIM, HDIM);
  transpose_bf16_k<<<dim3((2 * FFD) / 32, HDIM / 32, 2), blkT, 0, stream>>>(
      W_gu, wt_gu, HDIM, 2 * FFD);
  transpose_bf16_k<<<dim3(HDIM / 32, FFD / 32, 2), blkT, 0, stream>>>(
      W_d, wt_d, FFD, HDIM);

  // 1) h = rmsnorm(x, ln1) -> bf16
  rmsnorm_bf16_k<<<M_TOTAL, blk256, 0, stream>>>(x, ln1, sp, h_bf);
  // 2) qkv = h @ W_qkv[model]  (bf16 out)
  gemm_routed_k<true, false><<<dim3(M_TOTAL / BM, (3 * HDIM) / BN), blk256, 0,
                               stream>>>(h_bf, wt_qkv, nullptr, nullptr, qkv_bf,
                                         sp, 3 * HDIM, HDIM);
  // 3) flash attention per segment/head
  flash_attn_k<<<dim3(M_TOTAL / 64, NHEADS, 2), blk128, 0, stream>>>(qkv_bf,
                                                                     attn_bf, sp);
  // 4) x2 = x + attn @ W_o[model]
  gemm_routed_k<false, true><<<dim3(M_TOTAL / BM, HDIM / BN), blk256, 0,
                               stream>>>(attn_bf, wt_o, x, x2, nullptr, sp,
                                         HDIM, HDIM);
  // 5) h = rmsnorm(x2, ln2) -> bf16
  rmsnorm_bf16_k<<<M_TOTAL, blk256, 0, stream>>>(x2, ln2, sp, h_bf);
  // 6) gu = h @ W_gu[model]
  gemm_routed_k<false, false><<<dim3(M_TOTAL / BM, (2 * FFD) / BN), blk256, 0,
                                stream>>>(h_bf, wt_gu, nullptr, gu, nullptr, sp,
                                          2 * FFD, HDIM);
  // 7) hmlp = silu(gate)*up -> bf16
  {
    size_t total = (size_t)M_TOTAL * FFD;
    int blocks = (int)((total + 255) / 256);
    swiglu_k<<<blocks, blk256, 0, stream>>>(gu, hmlp_bf);
  }
  // 8) out = x2 + hmlp @ W_d[model]
  gemm_routed_k<false, true><<<dim3(M_TOTAL / BM, HDIM / BN), blk256, 0,
                               stream>>>(hmlp_bf, wt_d, x2, outf, nullptr, sp,
                                         HDIM, FFD);
}
